// Mamba2Mixer_58600533787160
// MI455X (gfx1250) — matrix operand order repaired and checked by reading, not compiled, not run on hardware
//
#include <hip/hip_runtime.h>
#include <hip/hip_bf16.h>

// ---------------------------------------------------------------------------
// Problem constants (Mamba2 mixer)
// ---------------------------------------------------------------------------
constexpr int cB    = 2;
constexpr int cS    = 4096;
constexpr int cHID  = 2048;
constexpr int cI    = 4096;
constexpr int cN    = 128;
constexpr int cH    = 64;
constexpr int cP    = 64;
constexpr int cK    = 4;
constexpr int cCH   = 256;
constexpr int cNC   = cS / cCH;             // 16
constexpr int cCONV = cI + 2 * cN;          // 4352
constexpr int cPROJ = 2 * (cI + cN) + cH;   // 8512
constexpr float cDT_MAX = 100.0f;
constexpr float cEPS    = 1e-5f;

constexpr int cROWS = cB * cS;              // 8192

// ---------------------------------------------------------------------------
// Workspace layout (float elements)
// ---------------------------------------------------------------------------
constexpr size_t OFF_ZX     = 0;                                        // (ROWS, PROJ)
constexpr size_t SZ_ZX      = (size_t)cROWS * cPROJ;
constexpr size_t OFF_CONV   = OFF_ZX + SZ_ZX;                           // (ROWS, CONV)
constexpr size_t SZ_CONV    = (size_t)cROWS * cCONV;
constexpr size_t OFF_DT     = OFF_CONV + SZ_CONV;                       // (ROWS, H)
constexpr size_t SZ_DT      = (size_t)cROWS * cH;
constexpr size_t OFF_DA     = OFF_DT + SZ_DT;                           // (ROWS, H)
constexpr size_t OFF_DACS   = OFF_DA + SZ_DT;                           // (ROWS, H)
constexpr size_t OFF_CDEC   = OFF_DACS + SZ_DT;                         // (B,NC,H)
constexpr size_t SZ_CDEC    = (size_t)cB * cNC * cH;
constexpr size_t OFF_STATES = OFF_CDEC + SZ_CDEC;                       // (B,NC,H,P,N) in-place -> prev
constexpr size_t SZ_STATES  = (size_t)cB * cNC * cH * cP * cN;
constexpr size_t OFF_Y      = OFF_STATES + SZ_STATES;                   // (ROWS, I), normalized in-place
constexpr size_t SZ_Y       = (size_t)cROWS * cI;

// ---------------------------------------------------------------------------
// WMMA helpers (wave32, 16x16x32 bf16, fp32 accum)
// ---------------------------------------------------------------------------
typedef __bf16 bf16_t;
typedef __attribute__((ext_vector_type(16))) __bf16 v16bf;
typedef __attribute__((ext_vector_type(8)))  float  v8f;

__device__ __forceinline__ int lane_id() { return (int)(threadIdx.x & 31); }

__device__ __forceinline__ v8f v8f_zero() {
  v8f z;
#pragma unroll
  for (int i = 0; i < 8; ++i) z[i] = 0.0f;
  return z;
}

// A fragment, 16x32 (MxK). src points at [m0][k0], row-major, leading dim ld.
// Lanes 0-15 -> M=lane, K=0..15; lanes 16-31 -> K=16..31 (matches B fragments).
__device__ __forceinline__ v16bf lds_a_frag(const bf16_t* s, int ld) {
  const int l  = lane_id();
  const int m  = l & 15;
  const int kb = (l >> 4) << 4;   // 0 or 16
  v16bf f;
#pragma unroll
  for (int j = 0; j < 8; ++j) {
    const int k = kb + 2 * j;
    f[2 * j]     = s[m * ld + k];
    f[2 * j + 1] = s[m * ld + k + 1];
  }
  return f;
}

// Same A-fragment layout, but staging is fp32 in LDS; convert on load.
__device__ __forceinline__ v16bf lds_a_frag_f32(const float* s, int ld) {
  const int l  = lane_id();
  const int m  = l & 15;
  const int kb = (l >> 4) << 4;
  v16bf f;
#pragma unroll
  for (int j = 0; j < 8; ++j) {
    const int k = kb + 2 * j;
    f[2 * j]     = (bf16_t)s[m * ld + k];
    f[2 * j + 1] = (bf16_t)s[m * ld + k + 1];
  }
  return f;
}

// B fragment, 32x16 (KxN), element (k,n) = src[k*ld + n] (k-major staging).
__device__ __forceinline__ v16bf lds_b_frag(const bf16_t* s, int ld) {
  const int l  = lane_id();
  const int n  = l & 15;
  const int kb = (l >> 4) << 4;   // 0 or 16
  v16bf f;
#pragma unroll
  for (int j = 0; j < 8; ++j) {
    f[2 * j]     = s[(kb + 2 * j) * ld + n];
    f[2 * j + 1] = s[(kb + 2 * j + 1) * ld + n];
  }
  return f;
}

// B fragment where staging is n-major bf16: element (k,n) = src[n*ld + k].
__device__ __forceinline__ v16bf lds_bT_frag(const bf16_t* s, int ld) {
  const int l  = lane_id();
  const int n  = l & 15;
  const int kb = (l >> 4) << 4;
  v16bf f;
#pragma unroll
  for (int j = 0; j < 8; ++j) {
    f[2 * j]     = s[n * ld + kb + 2 * j];
    f[2 * j + 1] = s[n * ld + kb + 2 * j + 1];
  }
  return f;
}

// B fragment, n-major fp32 staging (convert on load).
__device__ __forceinline__ v16bf lds_bT_frag_f32(const float* s, int ld) {
  const int l  = lane_id();
  const int n  = l & 15;
  const int kb = (l >> 4) << 4;
  v16bf f;
#pragma unroll
  for (int j = 0; j < 8; ++j) {
    f[2 * j]     = (bf16_t)s[n * ld + kb + 2 * j];
    f[2 * j + 1] = (bf16_t)s[n * ld + kb + 2 * j + 1];
  }
  return f;
}

__device__ __forceinline__ v8f wmma_bf16(v16bf a, v16bf b, v8f c) {
  return __builtin_amdgcn_wmma_f32_16x16x32_bf16(false, a, false, b, (short)0, c,
                                                 false, false);
}

__device__ __forceinline__ float silu_f(float x) {
  return x / (1.0f + __expf(-x));
}

// Async DMA: 16 bytes global -> LDS per lane, tracked with ASYNCcnt.
// LDS flat addresses carry the wave-relative LDS byte offset in addr[31:0]
// (aperture bits live in [63:32], ISA 10.2), so truncation yields VDST.
__device__ __forceinline__ void async_g2l_b128(void* lds_dst, const float* gsrc) {
  const uint32_t loff = (uint32_t)(uintptr_t)lds_dst;
  asm volatile("global_load_async_to_lds_b128 %0, %1, off"
               :: "v"(loff), "v"(gsrc)
               : "memory");
}
__device__ __forceinline__ void wait_asynccnt_all() {
  asm volatile("s_wait_asynccnt 0x0" ::: "memory");
}
__device__ __forceinline__ void wait_asynccnt_12() {
  asm volatile("s_wait_asynccnt 0xc" ::: "memory");
}

// ---------------------------------------------------------------------------
// Kernel 1/9: C[M,N] = A[M,K] @ W[N,K]^T, fp32 in/out, bf16 WMMA core.
// Requirements: M % 128 == 0, N % 64 == 0, K % 32 == 0 (true for both calls).
// Block tile 128x64, 128 threads (4 waves), each wave a 32x64 strip.
// Double-buffered fp32 LDS tiles filled by GLOBAL_LOAD_ASYNC_TO_LDS_B128.
// ---------------------------------------------------------------------------
__global__ void gemm_bf16_nt(const float* __restrict__ A,
                             const float* __restrict__ W,
                             float* __restrict__ C,
                             int M, int N, int Kdim) {
  constexpr int BM = 128, BN = 64, BK = 32;
  __shared__ float As[2][BM * BK];   // 16 KB each
  __shared__ float Ws[2][BN * BK];   //  8 KB each

  const int m0   = blockIdx.x * BM;
  const int n0   = blockIdx.y * BN;
  const int tid  = (int)threadIdx.x;
  const int wave = tid >> 5;

  // per-thread async copy plan: A = 1024 16B segments, W = 512 segments.
  auto issue_tile = [&](int buf, int k0) {
#pragma unroll
    for (int i = 0; i < 8; ++i) {
      const int seg = tid + i * 128;          // 0..1023
      const int r = seg >> 3, cs4 = (seg & 7) * 4;
      async_g2l_b128(&As[buf][r * BK + cs4],
                     A + (size_t)(m0 + r) * Kdim + k0 + cs4);
    }
#pragma unroll
    for (int i = 0; i < 4; ++i) {
      const int seg = tid + i * 128;          // 0..511
      const int r = seg >> 3, cs4 = (seg & 7) * 4;
      async_g2l_b128(&Ws[buf][r * BK + cs4],
                     W + (size_t)(n0 + r) * Kdim + k0 + cs4);
    }
  };

  v8f acc[8];
#pragma unroll
  for (int i = 0; i < 8; ++i) acc[i] = v8f_zero();

  const int nK = Kdim / BK;
  issue_tile(0, 0);

  for (int kt = 0; kt < nK; ++kt) {
    const int cur = kt & 1;
    if (kt + 1 < nK) {
      issue_tile(cur ^ 1, (kt + 1) * BK);   // overlap next slab with compute
      wait_asynccnt_12();                   // current slab (issued earlier) done
    } else {
      wait_asynccnt_all();
    }
    __syncthreads();

    v16bf bfr[4];
#pragma unroll
    for (int nt = 0; nt < 4; ++nt)
      bfr[nt] = lds_bT_frag_f32(&Ws[cur][(nt * 16) * BK], BK);
#pragma unroll
    for (int mt = 0; mt < 2; ++mt) {
      const v16bf af = lds_a_frag_f32(&As[cur][(wave * 32 + mt * 16) * BK], BK);
#pragma unroll
      for (int nt = 0; nt < 4; ++nt)
        acc[mt * 4 + nt] = wmma_bf16(af, bfr[nt], acc[mt * 4 + nt]);
    }
    __syncthreads();   // readers done before next issue overwrites buffer
  }

  const int nl = lane_id() & 15;
  const int mh = (lane_id() >> 4) * 8;
#pragma unroll
  for (int mt = 0; mt < 2; ++mt) {
#pragma unroll
    for (int nt = 0; nt < 4; ++nt) {
#pragma unroll
      for (int j = 0; j < 8; ++j) {
        const int gm = m0 + wave * 32 + mt * 16 + mh + j;
        const int gn = n0 + nt * 16 + nl;
        C[(size_t)gm * N + gn] = acc[mt * 4 + nt][j];
      }
    }
  }
}

// ---------------------------------------------------------------------------
// Kernel 2/9: depthwise causal conv1d (K=4) + bias + SiLU over xBC channels.
// ---------------------------------------------------------------------------
__global__ void conv_silu_kernel(const float* __restrict__ zx,
                                 const float* __restrict__ conv_w,
                                 const float* __restrict__ conv_b,
                                 float* __restrict__ out) {
  const size_t idx = (size_t)blockIdx.x * blockDim.x + threadIdx.x;
  const size_t total = (size_t)cROWS * cCONV;
  if (idx >= total) return;
  const int    c  = (int)(idx % cCONV);
  const size_t bs = idx / cCONV;
  const int    s  = (int)(bs % cS);
  const size_t b  = bs / cS;

  float acc = conv_b[c];
#pragma unroll
  for (int j = 0; j < cK; ++j) {
    const int sp = s - (cK - 1) + j;
    if (sp >= 0)
      acc += zx[((size_t)b * cS + sp) * cPROJ + cI + c] * conv_w[c * cK + j];
  }
  out[idx] = silu_f(acc);
}

// ---------------------------------------------------------------------------
// Kernel 3/9: dt = clip(softplus(dt_raw + bias)), dA = -exp(A_log)*dt.
// ---------------------------------------------------------------------------
__global__ void dt_kernel(const float* __restrict__ zx,
                          const float* __restrict__ dt_bias,
                          const float* __restrict__ A_log,
                          float* __restrict__ dt_out,
                          float* __restrict__ dA_out) {
  const size_t idx = (size_t)blockIdx.x * blockDim.x + threadIdx.x;
  if (idx >= (size_t)cROWS * cH) return;
  const int    h  = (int)(idx % cH);
  const size_t bs = idx / cH;
  const float v  = zx[bs * cPROJ + cI + cCONV + h] + dt_bias[h];
  float sp = (v > 20.0f) ? v : log1pf(__expf(v));
  sp = fminf(fmaxf(sp, 0.0f), cDT_MAX);
  dt_out[idx] = sp;
  dA_out[idx] = -__expf(A_log[h]) * sp;
}

// ---------------------------------------------------------------------------
// Kernel 4/9: per-(b,chunk,h) inclusive cumsum of dA + chunk decay.
// ---------------------------------------------------------------------------
__global__ void cumsum_kernel(const float* __restrict__ dA,
                              float* __restrict__ dAcs,
                              float* __restrict__ cdec) {
  const int idx = blockIdx.x * blockDim.x + threadIdx.x;
  if (idx >= cB * cNC * cH) return;
  const int h = idx % cH;
  const int c = (idx / cH) % cNC;
  const int b = idx / (cH * cNC);
  float cs = 0.0f;
  for (int l = 0; l < cCH; ++l) {
    const size_t e = ((size_t)b * cS + c * cCH + l) * cH + h;
    cs += dA[e];
    dAcs[e] = cs;
  }
  cdec[idx] = __expf(cs);
}

// ---------------------------------------------------------------------------
// Kernel 5/9: chunk states. One block per (b,c,h):
//   states[p,n] = sum_l x[l,p]*dt[l]*exp(cs_last - cs[l]) * B[l,n]
// 64x128 output, K=256, WMMA.
// ---------------------------------------------------------------------------
__global__ void states_kernel(const float* __restrict__ conv,
                              const float* __restrict__ dt,
                              const float* __restrict__ dAcs,
                              float* __restrict__ states) {
  const int h = blockIdx.x & 63;
  const int c = (blockIdx.x >> 6) & 15;
  const int b = blockIdx.x >> 10;
  const int tid  = (int)threadIdx.x;   // 128 threads, 4 waves
  const int wave = tid >> 5;

  __shared__ bf16_t As[cP * 32];    // [p][l32]
  __shared__ bf16_t Bs[32 * cN];    // [l32][n]
  __shared__ float  wl[cCH];

  const size_t row0 = (size_t)b * cS + (size_t)c * cCH;
  const float cs_last = dAcs[(row0 + cCH - 1) * cH + h];
  for (int l = tid; l < cCH; l += 128) {
    const size_t e = (row0 + l) * cH + h;
    wl[l] = dt[e] * __expf(cs_last - dAcs[e]);
  }
  __syncthreads();

  v8f acc[8];
#pragma unroll
  for (int i = 0; i < 8; ++i) acc[i] = v8f_zero();

  for (int l0 = 0; l0 < cCH; l0 += 32) {
    for (int i = tid; i < cP * 32; i += 128) {
      const int p = i >> 5, ll = i & 31;
      const float xv = conv[(row0 + l0 + ll) * cCONV + h * cP + p];
      As[p * 32 + ll] = (bf16_t)(xv * wl[l0 + ll]);
    }
    for (int i = tid; i < 32 * cN; i += 128) {
      const int ll = i >> 7, n = i & 127;
      Bs[ll * cN + n] = (bf16_t)conv[(row0 + l0 + ll) * cCONV + cI + n];
    }
    __syncthreads();

    const v16bf af = lds_a_frag(&As[(wave * 16) * 32], 32);
#pragma unroll
    for (int nt = 0; nt < 8; ++nt) {
      const v16bf bfb = lds_b_frag(&Bs[nt * 16], cN);
      acc[nt] = wmma_bf16(af, bfb, acc[nt]);
    }
    __syncthreads();
  }

  const int nl = lane_id() & 15;
  const int mh = (lane_id() >> 4) * 8;
  float* st = states + (((size_t)(b * cNC + c) * cH + h) * cP) * cN;
#pragma unroll
  for (int nt = 0; nt < 8; ++nt) {
#pragma unroll
    for (int j = 0; j < 8; ++j) {
      const int p = wave * 16 + mh + j;
      const int n = nt * 16 + nl;
      st[(size_t)p * cN + n] = acc[nt][j];
    }
  }
}

// ---------------------------------------------------------------------------
// Kernel 6/9: inter-chunk scan, in-place (states -> prev).
// ---------------------------------------------------------------------------
__global__ void scan_kernel(float* __restrict__ states,
                            const float* __restrict__ cdec) {
  const int b = blockIdx.x / cH;
  const int h = blockIdx.x % cH;
  const int t = (int)threadIdx.x;   // 256
  float run[32];
#pragma unroll
  for (int i = 0; i < 32; ++i) run[i] = 0.0f;

  for (int c = 0; c < cNC; ++c) {
    const float dec = cdec[(b * cNC + c) * cH + h];
    float* base = states + ((size_t)(b * cNC + c) * cH + h) * cP * cN;
#pragma unroll
    for (int i = 0; i < 32; ++i) {
      const size_t e = (size_t)i * 256 + t;
      const float st = base[e];
      base[e] = run[i];                 // prev (state before this chunk)
      run[i] = run[i] * dec + st;
    }
  }
}

// ---------------------------------------------------------------------------
// Kernel 7/9: intra-chunk output. One block per (b,c,h,lb), lb = 64-row slab.
//  Y[l,p] = sum_{s<=l} (C·B^T)[l,s] exp(cs_l - cs_s) * x[s,p] dt[s]
//         + exp(cs_l) * sum_n C[l,n] prev[p,n]  +  D[h] x[l,p]
// ---------------------------------------------------------------------------
__global__ void ychunk_kernel(const float* __restrict__ conv,
                              const float* __restrict__ dt,
                              const float* __restrict__ dAcs,
                              const float* __restrict__ prev,
                              const float* __restrict__ Dp,
                              float* __restrict__ y) {
  const int lb = blockIdx.x & 3;
  const int h  = (blockIdx.x >> 2) & 63;
  const int c  = (blockIdx.x >> 8) & 15;
  const int b  = blockIdx.x >> 12;
  const int tid  = (int)threadIdx.x;   // 128 threads, 4 waves
  const int wave = tid >> 5;

  __shared__ bf16_t Ct[64 * cN];      // C rows of the l-slab           16 KB
  __shared__ bf16_t BP[cN * 64];      // B s-tile [s][n] / prev^T [n][p] 16 KB
  __shared__ bf16_t XT[64 * 64];      // xdt s-tile [s][p]               8 KB
  __shared__ bf16_t CBs[64 * 64];     // masked CB tile [l][s]           8 KB
  __shared__ float  csl[64], css[64], dts[64];

  const int l0 = lb * 64;
  const size_t row0 = (size_t)b * cS + (size_t)c * cCH;

  for (int i = tid; i < 64 * cN; i += 128) {
    const int l = i >> 7, n = i & 127;
    Ct[i] = (bf16_t)conv[(row0 + l0 + l) * cCONV + cI + cN + n];
  }
  for (int i = tid; i < 64; i += 128)
    csl[i] = dAcs[(row0 + l0 + i) * cH + h];
  __syncthreads();

  v8f yacc[4];
#pragma unroll
  for (int i = 0; i < 4; ++i) yacc[i] = v8f_zero();

  const int nl = lane_id() & 15;
  const int mh = (lane_id() >> 4) * 8;

  for (int st = 0; st <= lb; ++st) {
    const int s0 = st * 64;
    for (int i = tid; i < 64; i += 128) {
      const size_t e = (row0 + s0 + i) * cH + h;
      css[i] = dAcs[e];
      dts[i] = dt[e];
    }
    __syncthreads();
    for (int i = tid; i < 64 * cN; i += 128) {
      const int s = i >> 7, n = i & 127;
      BP[i] = (bf16_t)conv[(row0 + s0 + s) * cCONV + cI + n];
    }
    for (int i = tid; i < 64 * 64; i += 128) {
      const int s = i >> 6, p = i & 63;
      XT[i] = (bf16_t)(conv[(row0 + s0 + s) * cCONV + h * cP + p] * dts[s]);
    }
    __syncthreads();

    // CB = C @ B^T over n (K = 128)
    v8f cb[4];
#pragma unroll
    for (int i = 0; i < 4; ++i) cb[i] = v8f_zero();
#pragma unroll
    for (int kk = 0; kk < 4; ++kk) {
      const v16bf af = lds_a_frag(&Ct[(wave * 16) * cN + kk * 32], cN);
#pragma unroll
      for (int nt = 0; nt < 4; ++nt) {
        const v16bf bfb = lds_bT_frag(&BP[(nt * 16) * cN + kk * 32], cN);
        cb[nt] = wmma_bf16(af, bfb, cb[nt]);
      }
    }
    // causal mask + decay scale, restage as bf16 A operand
#pragma unroll
    for (int nt = 0; nt < 4; ++nt) {
#pragma unroll
      for (int j = 0; j < 8; ++j) {
        const int l_loc = wave * 16 + mh + j;
        const int s_loc = nt * 16 + nl;
        const int gl = l0 + l_loc, gs = s0 + s_loc;
        const float sc = (gs <= gl) ? __expf(csl[l_loc] - css[s_loc]) : 0.0f;
        CBs[l_loc * 64 + s_loc] = (bf16_t)(cb[nt][j] * sc);
      }
    }
    // Y += (CB ∘ L) @ xdt  (K = 64); each wave reads only rows it wrote.
#pragma unroll
    for (int kk = 0; kk < 2; ++kk) {
      const v16bf af2 = lds_a_frag(&CBs[(wave * 16) * 64 + kk * 32], 64);
#pragma unroll
      for (int pt = 0; pt < 4; ++pt) {
        const v16bf bf2 = lds_b_frag(&XT[(kk * 32) * 64 + pt * 16], 64);
        yacc[pt] = wmma_bf16(af2, bf2, yacc[pt]);
      }
    }
    __syncthreads();
  }

  // Y_off: stage prev^T into BP as [n][p], GEMM over n (K = 128)
  const float* pv = prev + ((size_t)(b * cNC + c) * cH + h) * cP * cN;
  for (int i = tid; i < cN * 64; i += 128) {
    const int n = i >> 6, p = i & 63;
    BP[n * 64 + p] = (bf16_t)pv[(size_t)p * cN + n];
  }
  __syncthreads();

  v8f oacc[4];
#pragma unroll
  for (int i = 0; i < 4; ++i) oacc[i] = v8f_zero();
#pragma unroll
  for (int kk = 0; kk < 4; ++kk) {
    const v16bf af = lds_a_frag(&Ct[(wave * 16) * cN + kk * 32], cN);
#pragma unroll
    for (int pt = 0; pt < 4; ++pt) {
      const v16bf bfb = lds_b_frag(&BP[(kk * 32) * 64 + pt * 16], 64);
      oacc[pt] = wmma_bf16(af, bfb, oacc[pt]);
    }
  }

  const float Dh = Dp[h];
#pragma unroll
  for (int pt = 0; pt < 4; ++pt) {
#pragma unroll
    for (int j = 0; j < 8; ++j) {
      const int l_loc = wave * 16 + mh + j;
      const int p     = pt * 16 + nl;
      const size_t row = row0 + l0 + l_loc;
      const float xv = conv[row * cCONV + h * cP + p];
      const float v = yacc[pt][j] + __expf(csl[l_loc]) * oacc[pt][j] + Dh * xv;
      y[row * cI + h * cP + p] = v;
    }
  }
}

// ---------------------------------------------------------------------------
// Kernel 8/9: gated SiLU + RMSNorm, in-place on y. One block per token row.
// ---------------------------------------------------------------------------
__global__ void norm_kernel(float* __restrict__ y,
                            const float* __restrict__ zx,
                            const float* __restrict__ norm_w) {
  const size_t row = blockIdx.x;
  const int tid = (int)threadIdx.x;   // 256
  __shared__ float red[256];

  float vals[16];
  float local = 0.0f;
#pragma unroll
  for (int j = 0; j < 16; ++j) {
    const int i = tid + j * 256;
    const float yv = y[row * cI + i];
    const float zv = zx[row * cPROJ + i];
    const float g = yv * silu_f(zv);
    vals[j] = g;
    local += g * g;
  }
  red[tid] = local;
  __syncthreads();
  for (int off = 128; off > 0; off >>= 1) {
    if (tid < off) red[tid] += red[tid + off];
    __syncthreads();
  }
  const float inv = rsqrtf(red[0] / (float)cI + cEPS);
#pragma unroll
  for (int j = 0; j < 16; ++j) {
    const int i = tid + j * 256;
    y[row * cI + i] = norm_w[i] * vals[j] * inv;
  }
}

// ---------------------------------------------------------------------------
// Launch
// ---------------------------------------------------------------------------
extern "C" void kernel_launch(void* const* d_in, const int* in_sizes, int n_in,
                              void* d_out, int out_size, void* d_ws, size_t ws_size,
                              hipStream_t stream) {
  (void)in_sizes; (void)n_in; (void)out_size; (void)ws_size;

  const float* hidden     = (const float*)d_in[0];
  const float* in_proj_w  = (const float*)d_in[1];
  const float* conv_w     = (const float*)d_in[2];
  const float* conv_b     = (const float*)d_in[3];
  const float* dt_bias    = (const float*)d_in[4];
  const float* A_log      = (const float*)d_in[5];
  const float* Dp         = (const float*)d_in[6];
  const float* norm_w     = (const float*)d_in[7];
  const float* out_proj_w = (const float*)d_in[8];
  float* out = (float*)d_out;

  float* ws     = (float*)d_ws;
  float* zx     = ws + OFF_ZX;
  float* conv   = ws + OFF_CONV;
  float* dt     = ws + OFF_DT;
  float* dA     = ws + OFF_DA;
  float* dAcs   = ws + OFF_DACS;
  float* cdec   = ws + OFF_CDEC;
  float* states = ws + OFF_STATES;   // becomes `prev` after scan
  float* y      = ws + OFF_Y;

  // 1) in_proj: (ROWS,PROJ) = hidden(ROWS,HID) @ in_proj_w(PROJ,HID)^T
  gemm_bf16_nt<<<dim3(cROWS / 128, cPROJ / 64), 128, 0, stream>>>(
      hidden, in_proj_w, zx, cROWS, cPROJ, cHID);

  // 2) depthwise conv + SiLU
  {
    const size_t total = (size_t)cROWS * cCONV;
    conv_silu_kernel<<<(unsigned)((total + 255) / 256), 256, 0, stream>>>(
        zx, conv_w, conv_b, conv);
  }

  // 3) dt / dA
  {
    const size_t total = (size_t)cROWS * cH;
    dt_kernel<<<(unsigned)((total + 255) / 256), 256, 0, stream>>>(
        zx, dt_bias, A_log, dt, dA);
  }

  // 4) chunk cumsum + chunk decay
  cumsum_kernel<<<(cB * cNC * cH + 255) / 256, 256, 0, stream>>>(dA, dAcs, cdec);

  // 5) chunk states (WMMA)
  states_kernel<<<cB * cNC * cH, 128, 0, stream>>>(conv, dt, dAcs, states);

  // 6) inter-chunk scan (in-place states -> prev)
  scan_kernel<<<cB * cH, 256, 0, stream>>>(states, cdec);

  // 7) intra-chunk outputs (WMMA x3)
  ychunk_kernel<<<cB * cNC * cH * 4, 128, 0, stream>>>(
      conv, dt, dAcs, states, Dp, y);

  // 8) gated RMSNorm (in-place on y)
  norm_kernel<<<cROWS, 256, 0, stream>>>(y, zx, norm_w);

  // 9) out_proj: (ROWS,HID) = y(ROWS,I) @ out_proj_w(HID,I)^T
  gemm_bf16_nt<<<dim3(cROWS / 128, cHID / 64), 128, 0, stream>>>(
      y, out_proj_w, out, cROWS, cHID, cI);
}
